// TSP_Decoder_74964359185045
// MI455X (gfx1250) — matrix operand order repaired and checked by reading, not compiled, not run on hardware
//
// TSP decoder for MI455X (gfx1250, wave32, WMMA bf16 16x16x32 everywhere).
// Memory-bound problem (~250MB min traffic @ 23.3TB/s ~ 11us; 15.6 GFLOP is
// negligible at bf16-WMMA rates), so: bf16 operands, f32 accumulate, flash
// attention with online softmax, final softmax two-pass with the raw-score
// slab resident in the 192MB L2.
#include <hip/hip_runtime.h>
#include <hip/hip_bf16.h>
#include <cmath>

#define Bsz   16
#define Pn    1000
#define Nn    1000
#define En    128
#define Hh    8
#define Dd    16
#define NPAD  1024
#define PTILES 63   // ceil(1000/16)

typedef __attribute__((ext_vector_type(16))) __bf16 v16bf;
typedef __attribute__((ext_vector_type(8)))  float  v8f;
typedef __attribute__((ext_vector_type(4)))  unsigned int v4u;
typedef __attribute__((ext_vector_type(4)))  float  v4f;

union BF16x16 { v16bf v; v4u q[2]; __bf16 e[16]; };
union BF16x8  { v4u q; __bf16 e[8]; };

__device__ __forceinline__ v4u zero4() { v4u z = {0u,0u,0u,0u}; return z; }
__device__ __forceinline__ v8f zero8() {
  v8f z;
#pragma unroll
  for (int i = 0; i < 8; ++i) z[i] = 0.0f;
  return z;
}

__device__ __forceinline__ v8f wmma_bf16(v16bf a, v16bf b, v8f c) {
  // v_wmma_f32_16x16x32_bf16: (neg_a, A, neg_b, B, c_mod, C, reuse_a, reuse_b)
  return __builtin_amdgcn_wmma_f32_16x16x32_bf16(false, a, false, b, (short)0, c,
                                                 false, false);
}

// A-operand (16x32 bf16) built from an f32 row: lane holds row m = lane%16,
// K pattern {8kh..8kh+7, 16+8kh..16+8kh+7} with kh = lane/16.
__device__ __forceinline__ v16bf load_a_f32(const float* rowBase, int kOff, int kh,
                                            bool valid) {
  BF16x16 r;
  if (valid) {
    const float* p = rowBase + kOff + kh * 8;
    v4f x0 = *(const v4f*)(p + 0);
    v4f x1 = *(const v4f*)(p + 4);
    v4f y0 = *(const v4f*)(p + 16);
    v4f y1 = *(const v4f*)(p + 20);
    r.e[0]  = (__bf16)x0.x; r.e[1]  = (__bf16)x0.y; r.e[2]  = (__bf16)x0.z; r.e[3]  = (__bf16)x0.w;
    r.e[4]  = (__bf16)x1.x; r.e[5]  = (__bf16)x1.y; r.e[6]  = (__bf16)x1.z; r.e[7]  = (__bf16)x1.w;
    r.e[8]  = (__bf16)y0.x; r.e[9]  = (__bf16)y0.y; r.e[10] = (__bf16)y0.z; r.e[11] = (__bf16)y0.w;
    r.e[12] = (__bf16)y1.x; r.e[13] = (__bf16)y1.y; r.e[14] = (__bf16)y1.z; r.e[15] = (__bf16)y1.w;
  } else {
    r.q[0] = zero4(); r.q[1] = zero4();
  }
  return r.v;
}

// B-operand (32x16 bf16): lane holds column n = lane%16, K pattern
// {8kh..8kh+7, 16+8kh..16+8kh+7}; p points at row 8kh, column n (row stride En).
__device__ __forceinline__ v16bf load_b_bf16(const __bf16* p) {
  BF16x16 r;
#pragma unroll
  for (int s = 0; s < 8; ++s) { r.e[s] = p[s * En]; r.e[8 + s] = p[(16 + s) * En]; }
  return r.v;
}

// Cross-lane reduction within each 16-lane half (rows of the C layout).
__device__ __forceinline__ float half16_max(float x) {
#pragma unroll
  for (int o = 1; o < 16; o <<= 1) x = fmaxf(x, __shfl_xor(x, o, 32));
  return x;
}
__device__ __forceinline__ float half16_sum(float x) {
#pragma unroll
  for (int o = 1; o < 16; o <<= 1) x += __shfl_xor(x, o, 32);
  return x;
}

// ---------------------------------------------------------------- kernels ---

// Convert the five 128x128 weight matrices to bf16 (packed [Wqf|Wql|Wk|Wv|Wmhc]).
__global__ void cvt_weights_kernel(const float* w0, const float* w1, const float* w2,
                                   const float* w3, const float* w4, __bf16* dst) {
  int i = blockIdx.x * blockDim.x + threadIdx.x;
  if (i >= 5 * En * En) return;
  int mi = i >> 14, ei = i & (En * En - 1);
  const float* s = (mi == 0) ? w0 : (mi == 1) ? w1 : (mi == 2) ? w2 : (mi == 3) ? w3 : w4;
  dst[i] = (__bf16)s[ei];
}

// Per-batch mean over nodes, then qmean[b] = mean @ Wq_mean  (tiny GEMV).
__global__ void mean_q_kernel(const float* enc, const float* Wqm, float* qmean) {
  int b = blockIdx.x;
  int j = threadIdx.x;                 // 128 threads
  __shared__ float mv[En];
  float s = 0.0f;
  for (int n = 0; n < Nn; ++n) s += enc[((size_t)b * Nn + n) * En + j];
  mv[j] = s * (1.0f / (float)Nn);
  __syncthreads();
  float acc = 0.0f;
#pragma unroll 8
  for (int k = 0; k < En; ++k) acc += mv[k] * Wqm[k * En + j];
  qmean[b * En + j] = acc;
}

// encT[b][e][n] = bf16(enc[b][n][e]), zero-padded to NPAD columns.
__global__ void transpose_enc_kernel(const float* enc, __bf16* encT) {
  int i = blockIdx.x * blockDim.x + threadIdx.x;   // 16*128*1024 exactly
  int n = i & (NPAD - 1);
  int e = (i >> 10) & (En - 1);
  int b = i >> 17;
  float v = (n < Nn) ? enc[((size_t)b * Nn + n) * En + e] : 0.0f;
  encT[((size_t)b * En + e) * NPAD + n] = (__bf16)v;
}

// Q = first@Wqf + last@Wql + qmean, stored bf16 [B][NPAD][128], padded rows = qmean alone.
__global__ void proj_q_kernel(const float* firstN, const float* lastN, const float* qmean,
                              const __bf16* wqf, const __bf16* wql, __bf16* Qbf) {
  int lane = threadIdx.x & 31;
  int wave = (blockIdx.x * blockDim.x + threadIdx.x) >> 5;
  int jt = wave & 7, pt = (wave >> 3) & 63, b = wave >> 9;
  if (b >= Bsz) return;
  int m = lane & 15, kh = lane >> 4;
  int prow = pt * 16 + m;
  bool valid = prow < Pn;
  const float* fr = firstN + ((size_t)b * Pn + prow) * En;
  const float* lr = lastN  + ((size_t)b * Pn + prow) * En;
  v8f acc = zero8();
#pragma unroll
  for (int kk = 0; kk < 4; ++kk) {
    v16bf af = load_a_f32(fr, kk * 32, kh, valid);
    v16bf al = load_a_f32(lr, kk * 32, kh, valid);
    int krow = kk * 32 + 8 * kh;
    acc = wmma_bf16(af, load_b_bf16(wqf + krow * En + jt * 16 + m), acc);
    acc = wmma_bf16(al, load_b_bf16(wql + krow * En + jt * 16 + m), acc);
  }
  float qm = qmean[b * En + jt * 16 + m];
#pragma unroll
  for (int r = 0; r < 8; ++r) {
    int p = pt * 16 + r + 8 * kh;
    Qbf[((size_t)b * NPAD + p) * En + jt * 16 + m] = (__bf16)(acc[r] + qm);
  }
}

// K and V projections. K stored transposed per head: Kt[b][h][d][NPAD] (bf16, rows
// contiguous over n -> perfect B-operand). V stored [b][NPAD][128] bf16.
__global__ void proj_kv_kernel(const float* enc, const __bf16* wk, const __bf16* wv,
                               __bf16* Kt, __bf16* Vbf) {
  int lane = threadIdx.x & 31;
  int wave = (blockIdx.x * blockDim.x + threadIdx.x) >> 5;
  int jt = wave & 7, nt = (wave >> 3) & 63, b = (wave >> 9) & 15, sel = wave >> 13;
  int m = lane & 15, kh = lane >> 4;
  int nrow = nt * 16 + m;
  bool valid = nrow < Nn;
  const float* ar = enc + ((size_t)b * Nn + nrow) * En;
  const __bf16* W = sel ? wv : wk;
  v8f acc = zero8();
#pragma unroll
  for (int kk = 0; kk < 4; ++kk) {
    v16bf a = load_a_f32(ar, kk * 32, kh, valid);
    int krow = kk * 32 + 8 * kh;
    acc = wmma_bf16(a, load_b_bf16(W + krow * En + jt * 16 + m), acc);
  }
  if (sel == 0) {
    int j = jt * 16 + m, h = j >> 4, d = j & 15;
    BF16x8 pk;
#pragma unroll
    for (int r = 0; r < 8; ++r) pk.e[r] = (__bf16)acc[r];   // 8 consecutive n values
    *(v4u*)(Kt + (((size_t)b * Hh + h) * 16 + d) * NPAD + nt * 16 + 8 * kh) = pk.q;
  } else {
#pragma unroll
    for (int r = 0; r < 8; ++r) {
      int n = nt * 16 + r + 8 * kh;
      Vbf[((size_t)b * NPAD + n) * En + jt * 16 + m] = (__bf16)acc[r];
    }
  }
}

// Flash attention: one wave per (b, h, 16-row P tile). Online softmax over N
// in 32-column chunks; probs re-staged via per-wave LDS tile for the PV WMMA.
__global__ void attn_kernel(const float* ninf, const __bf16* Qbf, const __bf16* Kt,
                            const __bf16* Vbf, float* attn) {
  int lane = threadIdx.x & 31;
  int wv = threadIdx.x >> 5;
  int wave = blockIdx.x * (blockDim.x >> 5) + wv;
  int pt = wave % PTILES;
  int t = wave / PTILES;
  int h = t & 7, b = t >> 3;
  if (b >= Bsz) return;
  __shared__ __attribute__((aligned(16))) __bf16 pbuf[4][16 * 32];
  __bf16* pb = pbuf[wv];
  int m = lane & 15, kh = lane >> 4;
  int p0 = pt * 16;

  BF16x16 aq;  // Q tile: K-dim 0..15 real (head dims), 16..31 zero.
  aq.q[0] = *(const v4u*)(Qbf + ((size_t)b * NPAD + p0 + m) * En + h * 16 + kh * 8);
  aq.q[1] = zero4();

  float mrow[8], lrow[8];
#pragma unroll
  for (int r = 0; r < 8; ++r) { mrow[r] = -3.0e38f; lrow[r] = 0.0f; }
  v8f acc = zero8();
  const __bf16* ktbase = Kt + (((size_t)b * Hh + h) * 16) * NPAD;

  for (int n0 = 0; n0 < NPAD; n0 += 32) {
    BF16x16 bk0, bk1;
    {  // B k = d (8kh..8kh+7) from K^T at column n; k 16..31 are the zero K half
      const __bf16* kp = ktbase + (size_t)(8 * kh) * NPAD + n0 + m;
#pragma unroll
      for (int s = 0; s < 8; ++s) { bk0.e[s] = kp[(size_t)s * NPAD]; bk1.e[s] = kp[(size_t)s * NPAD + 16]; }
      bk0.q[1] = zero4(); bk1.q[1] = zero4();
      __builtin_prefetch(kp + 32, 0, 1);
    }
    v8f s0 = wmma_bf16(aq.v, bk0.v, zero8());
    v8f s1 = wmma_bf16(aq.v, bk1.v, zero8());

#pragma unroll
    for (int r = 0; r < 8; ++r) {
      int p = p0 + r + 8 * kh;
      int ng0 = n0 + m, ng1 = n0 + 16 + m;
      float msk0 = (ng0 < Nn) ? ((p < Pn) ? ninf[((size_t)b * Pn + p) * Nn + ng0] : 0.0f)
                              : -1.0e30f;
      float msk1 = (ng1 < Nn) ? ((p < Pn) ? ninf[((size_t)b * Pn + p) * Nn + ng1] : 0.0f)
                              : -1.0e30f;
      float v0 = s0[r] * 0.25f + msk0;    // 1/sqrt(D), D=16
      float v1 = s1[r] * 0.25f + msk1;
      float rmax = half16_max(fmaxf(v0, v1));
      float mnew = fmaxf(mrow[r], rmax);
      float corr = __expf(mrow[r] - mnew);
      float e0 = __expf(v0 - mnew), e1 = __expf(v1 - mnew);
      lrow[r] = lrow[r] * corr + half16_sum(e0 + e1);
      mrow[r] = mnew;
      acc[r] *= corr;
      int rr = r + 8 * kh;
      pb[rr * 32 + m]      = (__bf16)e0;   // C layout -> LDS (row-major 16x32)
      pb[rr * 32 + 16 + m] = (__bf16)e1;
    }
    __builtin_amdgcn_wave_barrier();
    asm volatile("s_wait_dscnt 0" ::: "memory");

    BF16x16 ap, bv;   // reload probs in A layout, V chunk in B layout
    ap.q[0] = *(const v4u*)&pb[m * 32 + kh * 8];
    ap.q[1] = *(const v4u*)&pb[m * 32 + 16 + kh * 8];
    const __bf16* vp = Vbf + ((size_t)b * NPAD + n0 + 8 * kh) * En + h * 16 + m;
    bv.v = load_b_bf16(vp);
    __builtin_prefetch(vp + 32 * En, 0, 1);
    acc = wmma_bf16(ap.v, bv.v, acc);
    __builtin_amdgcn_wave_barrier();   // keep next iter's LDS stores after these loads
  }

#pragma unroll
  for (int r = 0; r < 8; ++r) {
    int p = p0 + r + 8 * kh;
    if (p < Pn)
      attn[((size_t)b * NPAD + p) * En + h * 16 + m] = acc[r] / lrow[r];
  }
}

// mh = attn @ W_mhc + b_mhc, stored bf16 [B][NPAD][128].
__global__ void outproj_kernel(const float* attn, const __bf16* wmhc, const float* bvec,
                               __bf16* mh) {
  int lane = threadIdx.x & 31;
  int wave = (blockIdx.x * blockDim.x + threadIdx.x) >> 5;
  int jt = wave & 7, pt = (wave >> 3) & 63, b = wave >> 9;
  if (b >= Bsz) return;
  int m = lane & 15, kh = lane >> 4;
  int prow = pt * 16 + m;
  bool valid = prow < Pn;
  const float* ar = attn + ((size_t)b * NPAD + prow) * En;
  v8f acc = zero8();
#pragma unroll
  for (int kk = 0; kk < 4; ++kk) {
    v16bf a = load_a_f32(ar, kk * 32, kh, valid);
    int krow = kk * 32 + 8 * kh;
    acc = wmma_bf16(a, load_b_bf16(wmhc + krow * En + jt * 16 + m), acc);
  }
  float bias = bvec[jt * 16 + m];
#pragma unroll
  for (int r = 0; r < 8; ++r) {
    int p = pt * 16 + r + 8 * kh;
    mh[((size_t)b * NPAD + p) * En + jt * 16 + m] = (__bf16)(acc[r] + bias);
  }
}

// score2 = mh @ enc^T / sqrt(E); 10*tanh + mask; softmax over N. Pass 1 writes
// raw masked scores to d_out while tracking row max/sum; pass 2 (same wave,
// L2-resident slab) normalizes in place.
__global__ void final_kernel(const float* ninf, const __bf16* mh, const __bf16* encT,
                             float* out) {
  int lane = threadIdx.x & 31;
  int wv = threadIdx.x >> 5;
  int wave = blockIdx.x * (blockDim.x >> 5) + wv;
  int pt = wave % PTILES;
  int b = wave / PTILES;
  if (b >= Bsz) return;
  __shared__ float stats[4][16][2];
  int m = lane & 15, kh = lane >> 4;
  int p0 = pt * 16;

  BF16x16 a[4];
  const __bf16* arow = mh + ((size_t)b * NPAD + p0 + m) * En;
#pragma unroll
  for (int kk = 0; kk < 4; ++kk) {
    a[kk].q[0] = *(const v4u*)(arow + kk * 32 + kh * 8);
    a[kk].q[1] = *(const v4u*)(arow + kk * 32 + 16 + kh * 8);
  }
  float mrow[8], lrow[8];
#pragma unroll
  for (int r = 0; r < 8; ++r) { mrow[r] = -3.0e38f; lrow[r] = 0.0f; }
  const float invSqrtE = 0.08838834764831845f;  // 1/sqrt(128)

  for (int n0 = 0; n0 < NPAD; n0 += 16) {
    v8f s = zero8();
#pragma unroll
    for (int kk = 0; kk < 4; ++kk) {
      const __bf16* brow = encT + ((size_t)b * En + kk * 32 + 8 * kh) * NPAD + n0 + m;
      BF16x16 bb;
#pragma unroll
      for (int s = 0; s < 8; ++s) { bb.e[s] = brow[(size_t)s * NPAD]; bb.e[8 + s] = brow[(size_t)(16 + s) * NPAD]; }
      s = wmma_bf16(a[kk].v, bb.v, s);
    }
    int n = n0 + m;
#pragma unroll
    for (int r = 0; r < 8; ++r) {
      int p = p0 + r + 8 * kh;
      float val;
      if (n < Nn) {
        float msk = (p < Pn) ? ninf[((size_t)b * Pn + p) * Nn + n] : 0.0f;
        val = 10.0f * tanhf(s[r] * invSqrtE) + msk;
      } else {
        val = -1.0e30f;
      }
      float rmax = half16_max(val);
      float mnew = fmaxf(mrow[r], rmax);
      float corr = __expf(mrow[r] - mnew);
      lrow[r] = lrow[r] * corr + half16_sum(__expf(val - mnew));
      mrow[r] = mnew;
      if (p < Pn && n < Nn) out[((size_t)b * Pn + p) * Nn + n] = val;
    }
  }
  if (lane == 0) {
#pragma unroll
    for (int r = 0; r < 8; ++r) { stats[wv][r][0] = mrow[r]; stats[wv][r][1] = lrow[r]; }
  }
  if (lane == 16) {
#pragma unroll
    for (int r = 0; r < 8; ++r) { stats[wv][8 + r][0] = mrow[r]; stats[wv][8 + r][1] = lrow[r]; }
  }
  __builtin_amdgcn_wave_barrier();
  asm volatile("s_wait_dscnt 0" ::: "memory");

  int rows = (Pn - p0) < 16 ? (Pn - p0) : 16;
  size_t base = ((size_t)b * Pn + p0) * Nn;
  int total = rows * Nn;
  for (int i = lane; i < total; i += 32) {
    int pr = i / Nn;
    float x = out[base + i];
    out[base + i] = __expf(x - stats[wv][pr][0]) / stats[wv][pr][1];
  }
}

// ------------------------------------------------------------------ launch ---

extern "C" void kernel_launch(void* const* d_in, const int* in_sizes, int n_in,
                              void* d_out, int out_size, void* d_ws, size_t ws_size,
                              hipStream_t stream) {
  const float* enc    = (const float*)d_in[0];
  const float* firstN = (const float*)d_in[1];
  const float* lastN  = (const float*)d_in[2];
  const float* ninf   = (const float*)d_in[3];
  const float* Wqf    = (const float*)d_in[4];
  const float* Wql    = (const float*)d_in[5];
  const float* Wqm    = (const float*)d_in[6];
  const float* Wk     = (const float*)d_in[7];
  const float* Wv     = (const float*)d_in[8];
  const float* Wmhc   = (const float*)d_in[9];
  const float* bmhc   = (const float*)d_in[10];
  float* out = (float*)d_out;

  // Workspace layout (all 256B aligned), total ~29.6 MB.
  char* ws = (char*)d_ws;
  __bf16* WBF   = (__bf16*)(ws + 0);         // 5 * 128*128 bf16 = 160 KB
  float*  QMEAN = (float*)(ws + 163840);     // 16*128 f32
  __bf16* QBF   = (__bf16*)(ws + 172032);    // [16][1024][128] bf16
  __bf16* KT    = (__bf16*)(ws + 4366336);   // [16][8][16][1024] bf16
  __bf16* VBF   = (__bf16*)(ws + 8560640);   // [16][1024][128] bf16
  float*  ATTN  = (float*)(ws + 12754944);   // [16][1024][128] f32
  __bf16* MH    = (__bf16*)(ws + 21143552);  // [16][1024][128] bf16
  __bf16* ENCT  = (__bf16*)(ws + 25337856);  // [16][128][1024] bf16

  cvt_weights_kernel<<<320, 256, 0, stream>>>(Wqf, Wql, Wk, Wv, Wmhc, WBF);
  mean_q_kernel<<<Bsz, 128, 0, stream>>>(enc, Wqm, QMEAN);
  transpose_enc_kernel<<<8192, 256, 0, stream>>>(enc, ENCT);
  proj_q_kernel<<<2048, 128, 0, stream>>>(firstN, lastN, QMEAN,
                                          WBF + 0, WBF + 16384, QBF);
  proj_kv_kernel<<<4096, 128, 0, stream>>>(enc, WBF + 2 * 16384, WBF + 3 * 16384,
                                           KT, VBF);
  attn_kernel<<<(Bsz * Hh * PTILES) / 4, 128, 0, stream>>>(ninf, QBF, KT, VBF, ATTN);
  outproj_kernel<<<2048, 128, 0, stream>>>(ATTN, WBF + 4 * 16384, bmhc, MH);
  final_kernel<<<(Bsz * PTILES) / 4, 128, 0, stream>>>(ninf, MH, ENCT, out);
}
